// QCNN_simple_1bit_71141838291244
// MI455X (gfx1250) — compile-verified, matrix operand order checked by reading, not run on hardware
//
#include <hip/hip_runtime.h>
#include <hip/hip_bf16.h>
#include <math.h>

typedef __attribute__((ext_vector_type(8))) int v8i;
typedef __attribute__((ext_vector_type(2))) int v2i;

#ifndef __has_builtin
#define __has_builtin(x) 0
#endif

#if __has_builtin(__builtin_amdgcn_global_load_async_to_lds_b64)
#define USE_ASYNC 1
#else
#define USE_ASYNC 0
#endif

// async-copy pointer types: global (AS1) / LDS (AS3) pointers to 64-bit v2i
typedef __attribute__((address_space(1))) v2i as1_v2i;
typedef __attribute__((address_space(3))) v2i as3_v2i;

#define B_    32
#define H1    222
#define W1D   222
#define H2    220
#define W2D   220
#define C1    32
#define C2    64
#define KPIX  (H2*W2D)     // 48400 pixels per image
#define KTOT  288          // 9 taps * 32 channels
#define KPAD  320          // padded K (5 x 64)
#define ROWB  KPAD         // bytes per im2col row (u8)

// ---------------- workspace layout ----------------
// [0,16)      : 4 floats  : sW1, sW2, sWf1, sWf2
// [64,8256)   : 2048 uint : pooled bit counts [B][C2]
// [8256,..)   : 5120 int  : packed W2 sign fragments (IU8 B layout)
// [32768,..)  : u8        : y1 activations NHWC [B][222][222][32] (~48 MB)

// =====================================================================
// prep: weight scales, zero pool, pack sign(W2) into WMMA B fragments
// =====================================================================
__global__ void prep_kernel(const float* __restrict__ W1, const float* __restrict__ W2,
                            const float* __restrict__ Wf1, const float* __restrict__ Wf2,
                            float* __restrict__ scales, unsigned int* __restrict__ pool,
                            int* __restrict__ packB)
{
    __shared__ float red[256];
    int t = threadIdx.x;

    const float* ptrs[4] = {W1, W2, Wf1, Wf2};
    const int    ns[4]   = {32*3*3*3, 64*32*3*3, 128*64, 10*128};
    for (int m = 0; m < 4; ++m) {
        float s = 0.f;
        for (int i = t; i < ns[m]; i += 256) s += fabsf(ptrs[m][i]);
        red[t] = s; __syncthreads();
        for (int off = 128; off > 0; off >>= 1) {
            if (t < off) red[t] += red[t + off];
            __syncthreads();
        }
        if (t == 0) scales[m] = red[0] / (float)ns[m];
        __syncthreads();
    }

    for (int i = t; i < B_*C2; i += 256) pool[i] = 0u;

    // B fragment layout (mirrors 8-bit A layout, ISA §7.12.2):
    // packB[((nt*5 + s)*32 + lane)*8 + v]; each dword packs 4 K-consecutive i8.
    for (int idx = t; idx < 4*5*32*8; idx += 256) {
        int v    = idx & 7;
        int lane = (idx >> 3) & 31;
        int rest = idx >> 8;           // nt*5 + s
        int s    = rest % 5;
        int nt   = rest / 5;
        int nloc = lane & 15;
        int g    = lane >> 4;
        int kd   = ((v >> 1) << 2) + (v & 1) + (g << 1);
        int k0   = s * 64 + kd * 4;
        unsigned int w = 0;
        for (int j = 0; j < 4; ++j) {
            int k = k0 + j;
            int sv = 0;
            if (k < KTOT) {
                int tap = k >> 5;          // K = tap*32 + ic
                int ic  = k & 31;
                int ky  = tap / 3, kx = tap % 3;
                float wv = W2[(((nt*16 + nloc)*32 + ic)*3 + ky)*3 + kx];
                sv = (wv > 0.f) ? 1 : ((wv < 0.f) ? -1 : 0);
            }
            w |= ((unsigned int)(unsigned char)(signed char)sv) << (8*j);
        }
        packB[idx] = (int)w;
    }
}

// =====================================================================
// conv1 + qrelu -> u8 {0,1} activations, NHWC (32B contiguous channels)
// =====================================================================
__global__ void conv1_kernel(const float* __restrict__ x, const float* __restrict__ W1,
                             const float* __restrict__ b1, const float* __restrict__ s1p,
                             const float* __restrict__ scales, unsigned char* __restrict__ y1)
{
    __shared__ float wsgn[864];
    __shared__ float bsh[32];
    int t = threadIdx.x;
    for (int i = t; i < 864; i += 256) {
        float w = W1[i];
        wsgn[i] = (w > 0.f) ? 1.f : ((w < 0.f) ? -1.f : 0.f);
    }
    if (t < 32) bsh[t] = b1[t];
    __syncthreads();

    int gid = blockIdx.x * 256 + t;
    const int npix = H1 * W1D;
    if (gid >= B_ * npix) return;
    int b  = gid / npix;
    int p  = gid - b * npix;
    int oy = p / W1D;
    int ox = p - oy * W1D;

    float patch[27];
    #pragma unroll
    for (int ic = 0; ic < 3; ++ic)
        #pragma unroll
        for (int ky = 0; ky < 3; ++ky)
            #pragma unroll
            for (int kx = 0; kx < 3; ++kx)
                patch[ic*9 + ky*3 + kx] =
                    x[((size_t)(b*3 + ic)*224 + (oy + ky))*224 + (ox + kx)];

    float sW  = scales[0];
    float s1v = s1p[0];
    unsigned int outw[8];
    #pragma unroll
    for (int q = 0; q < 8; ++q) {
        unsigned int wq = 0;
        #pragma unroll
        for (int j = 0; j < 4; ++j) {
            int oc = q*4 + j;
            float acc = 0.f;
            #pragma unroll
            for (int k = 0; k < 27; ++k) acc = fmaf(wsgn[oc*27 + k], patch[k], acc);
            float y  = sW * acc + bsh[oc];
            float xc = fminf(fmaxf(y, 0.f), s1v);
            unsigned int bit = (unsigned int)rintf(xc / s1v);
            wq |= (bit & 1u) << (8*j);
        }
        outw[q] = wq;
    }
    uint4* dst = (uint4*)(y1 + (size_t)gid * 32);
    dst[0] = make_uint4(outw[0], outw[1], outw[2], outw[3]);
    dst[1] = make_uint4(outw[4], outw[5], outw[6], outw[7]);
}

// =====================================================================
// conv2 via v_wmma_i32_16x16x64_iu8 + fused qrelu + global-avg-pool count
// 8 waves/block; each wave owns a 16-pixel M tile, all 64 channels.
// B fragments live in VGPRs for the whole kernel (loaded once from L2);
// im2col staging uses async global->LDS copies (ASYNCcnt path).
// =====================================================================
__global__ __launch_bounds__(256) void conv2_kernel(
    const unsigned char* __restrict__ y1, const float* __restrict__ b2,
    const float* __restrict__ s1p, const float* __restrict__ s2p,
    const float* __restrict__ scales, const int* __restrict__ packB,
    unsigned int* __restrict__ pool)
{
    extern __shared__ char smem[];
    unsigned char* ldsA = (unsigned char*)smem;     // 8 waves * 16 rows * 320 B

    int t    = threadIdx.x;
    int lane = t & 31;
    int wave = t >> 5;
    int b    = blockIdx.y;

    // ---- preload all 20 B fragments (4 ntiles x 5 ksteps) into VGPRs ----
    v8i breg[20];
    #pragma unroll
    for (int f = 0; f < 20; ++f) {
        const int4* p = (const int4*)packB + (((f*32) + lane) << 1);
        int4 lo = p[0];
        int4 hi = p[1];
        breg[f][0] = lo.x; breg[f][1] = lo.y; breg[f][2] = lo.z; breg[f][3] = lo.w;
        breg[f][4] = hi.x; breg[f][5] = hi.y; breg[f][6] = hi.z; breg[f][7] = hi.w;
    }

    int pix0 = blockIdx.x * 128 + wave * 16;

    // ---- im2col rows: 9 contiguous 32B channel chunks, K padded to 320 ----
    for (int r = 0; r < 16; ++r) {
        int p = pix0 + r;
        bool valid = (p < KPIX);
        int oy = valid ? (p / W2D) : 0;
        int ox = valid ? (p - oy * W2D) : 0;
        unsigned char* rowp = ldsA + (size_t)(wave*16 + r) * ROWB;
        #pragma unroll
        for (int half = 0; half < 2; ++half) {
            int slot = lane + half*32;           // 40 x 8-byte slots per row
            if (slot < 40) {
                if (valid && slot < 36) {
                    int chunk = slot >> 2;       // tap 0..8
                    int off   = (slot & 3) * 8;
                    int ky = chunk / 3, kx = chunk - (chunk/3)*3;
                    const unsigned char* src =
                        y1 + ((size_t)(b*H1 + (oy + ky)) * W1D + (ox + kx)) * 32 + off;
#if USE_ASYNC
                    __builtin_amdgcn_global_load_async_to_lds_b64(
                        (as1_v2i*)src, (as3_v2i*)(rowp + slot*8), 0, 0);
#else
                    *(unsigned long long*)(rowp + slot*8) =
                        *(const unsigned long long*)src;
#endif
                } else {
                    *(unsigned long long*)(rowp + slot*8) = 0ull;
                }
            }
        }
    }
#if USE_ASYNC
  #if __has_builtin(__builtin_amdgcn_s_wait_asynccnt)
    __builtin_amdgcn_s_wait_asynccnt(0);
  #else
    asm volatile("s_wait_asynccnt 0x0" ::: "memory");
  #endif
#endif
    __syncthreads();

    // ---- A fragment gather (8-bit 16x64 layout, ISA §7.12.2) + WMMA ----
    const int2* arow2 =
        (const int2*)(ldsA + (size_t)(wave*16 + (lane & 15)) * ROWB);
    int gbit = lane >> 4;   // K-group select

    v8i acc[4] = {};
    #pragma unroll
    for (int s = 0; s < 5; ++s) {
        v8i a;
        #pragma unroll
        for (int q = 0; q < 4; ++q) {
            int2 pr = arow2[s*8 + gbit + q*2];   // ds_load_b64
            a[2*q]   = pr.x;
            a[2*q+1] = pr.y;
        }
        #pragma unroll
        for (int nt = 0; nt < 4; ++nt)
            acc[nt] = __builtin_amdgcn_wmma_i32_16x16x64_iu8(
                false, a, true, breg[nt*5 + s], acc[nt], false, false);
    }

    // ---- fused epilogue: scale+bias, 1-bit qrelu, avg-pool bit count ----
    float scale = scales[1] * s1p[0];
    float s2v   = s2p[0];
    int mbase   = (lane >> 4) * 8;   // C layout: lanes16-31 hold M+8
    int nloc    = lane & 15;
    #pragma unroll
    for (int nt = 0; nt < 4; ++nt) {
        int ch = nt*16 + nloc;
        float bias = b2[ch];
        unsigned int cnt = 0;
        #pragma unroll
        for (int v = 0; v < 8; ++v) {
            int p = pix0 + mbase + v;
            if (p < KPIX) {
                float y  = fmaf(scale, (float)acc[nt][v], bias);
                float xc = fminf(fmaxf(y, 0.f), s2v);
                cnt += (unsigned int)rintf(xc / s2v);
            }
        }
        atomicAdd(&pool[b*C2 + ch], cnt);
    }
}

// =====================================================================
// pooled -> fc1 (binary) -> qrelu -> fc2 (binary) -> log_softmax
// =====================================================================
__global__ void fc_kernel(const float* __restrict__ Wf1, const float* __restrict__ bf1,
                          const float* __restrict__ Wf2, const float* __restrict__ bf2,
                          const float* __restrict__ s2p, const float* __restrict__ s3p,
                          const float* __restrict__ scales,
                          const unsigned int* __restrict__ pool,
                          float* __restrict__ out)
{
    __shared__ float pooled[B_*C2];
    __shared__ float q[B_*128];
    __shared__ float logits[B_*10];
    int t = threadIdx.x;

    float s2v = s2p[0];
    for (int i = t; i < B_*C2; i += 320)
        pooled[i] = s2v * (float)pool[i] / (float)KPIX;
    __syncthreads();

    float sWf1 = scales[2], s3v = s3p[0];
    if (t < 128) {
        for (int b = 0; b < B_; ++b) {
            float acc = 0.f;
            for (int c = 0; c < C2; ++c) {
                float w  = Wf1[t*C2 + c];
                float sg = (w > 0.f) ? 1.f : ((w < 0.f) ? -1.f : 0.f);
                acc += sg * pooled[b*C2 + c];
            }
            float y  = sWf1 * acc + bf1[t];
            float xc = fminf(fmaxf(y, 0.f), s3v);
            q[b*128 + t] = s3v * rintf(xc / s3v);
        }
    }
    __syncthreads();

    float sWf2 = scales[3];
    if (t < B_*10) {
        int b = t / 10, o = t - b*10;
        float acc = 0.f;
        for (int k = 0; k < 128; ++k) {
            float w  = Wf2[o*128 + k];
            float sg = (w > 0.f) ? 1.f : ((w < 0.f) ? -1.f : 0.f);
            acc += sg * q[b*128 + k];
        }
        logits[t] = sWf2 * acc + bf2[o];
    }
    __syncthreads();

    if (t < B_) {
        int b = t;
        float m = logits[b*10];
        for (int o = 1; o < 10; ++o) m = fmaxf(m, logits[b*10 + o]);
        float se = 0.f;
        for (int o = 0; o < 10; ++o) se += expf(logits[b*10 + o] - m);
        float ls = logf(se) + m;
        for (int o = 0; o < 10; ++o) out[b*10 + o] = logits[b*10 + o] - ls;
    }
}

// =====================================================================
extern "C" void kernel_launch(void* const* d_in, const int* in_sizes, int n_in,
                              void* d_out, int out_size, void* d_ws, size_t ws_size,
                              hipStream_t stream)
{
    const float* x   = (const float*)d_in[0];
    const float* W1  = (const float*)d_in[1];
    const float* b1  = (const float*)d_in[2];
    const float* W2  = (const float*)d_in[3];
    const float* b2  = (const float*)d_in[4];
    const float* Wf1 = (const float*)d_in[5];
    const float* bf1 = (const float*)d_in[6];
    const float* Wf2 = (const float*)d_in[7];
    const float* bf2 = (const float*)d_in[8];
    const float* s1  = (const float*)d_in[9];
    const float* s2  = (const float*)d_in[10];
    const float* s3  = (const float*)d_in[11];

    char* ws = (char*)d_ws;
    float*        scales = (float*)(ws + 0);
    unsigned int* pool   = (unsigned int*)(ws + 64);
    int*          packB  = (int*)(ws + 64 + 8192);
    unsigned char* y1    = (unsigned char*)(ws + 32768);

    prep_kernel<<<1, 256, 0, stream>>>(W1, W2, Wf1, Wf2, scales, pool, packB);

    int totalPix1 = B_ * H1 * W1D;
    conv1_kernel<<<(totalPix1 + 255)/256, 256, 0, stream>>>(x, W1, b1, s1, scales, y1);

    dim3 g2((KPIX + 127)/128, B_);
    size_t lds2 = 8*16*ROWB;   // 40960 B (A tiles only)
    conv2_kernel<<<g2, 256, lds2, stream>>>(y1, b2, s1, s2, scales, packB, pool);

    fc_kernel<<<1, 320, 0, stream>>>(Wf1, bf1, Wf2, bf2, s2, s3, scales, pool,
                                     (float*)d_out);
}
